// RNN_Flat_27470610825310
// MI455X (gfx1250) — matrix operand order repaired and checked by reading, not compiled, not run on hardware
//
#include <hip/hip_runtime.h>

// Problem constants (from reference)
#define B_ 64
#define S_ 512
#define I_ 512
#define H_ 1024
#define L_ 6

typedef __attribute__((ext_vector_type(16))) __bf16 v16bf;
typedef __attribute__((ext_vector_type(8)))  __bf16 bf16x8;
typedef __attribute__((ext_vector_type(8)))  float  v8f;

__device__ __forceinline__ __bf16 f2bf(float f) {
    unsigned u = __builtin_bit_cast(unsigned, f);
    u += 0x7FFFu + ((u >> 16) & 1u);                 // round-to-nearest-even
    unsigned short h = (unsigned short)(u >> 16);
    return __builtin_bit_cast(__bf16, h);
}
__device__ __forceinline__ float bf2f(__bf16 b) {
    unsigned short h = __builtin_bit_cast(unsigned short, b);
    unsigned u = ((unsigned)h) << 16;
    return __builtin_bit_cast(float, u);
}
__device__ __forceinline__ v16bf cat16(bf16x8 lo, bf16x8 hi) {
    v16bf r;
#pragma unroll
    for (int i = 0; i < 8; ++i) { r[i] = lo[i]; r[i + 8] = hi[i]; }
    return r;
}

// ---- f32 -> bf16 bulk convert -------------------------------------------
__global__ void cvt_f32_bf16(const float* __restrict__ in, __bf16* __restrict__ out, int n) {
    for (int i = blockIdx.x * blockDim.x + threadIdx.x; i < n; i += gridDim.x * blockDim.x)
        out[i] = f2bf(in[i]);
}

// ---- zero fill (h state init) -------------------------------------------
__global__ void zero_u32(unsigned* __restrict__ p, int n) {
    for (int i = blockIdx.x * blockDim.x + threadIdx.x; i < n; i += gridDim.x * blockDim.x)
        p[i] = 0u;
}

// ---- WMMA GEMM with fused RNN epilogues ---------------------------------
// C[m,n] = sum_k A[m,k] * W[n,k]   (A: [M][K] bf16, W: [N][K] bf16, row-major)
// mode 0: f32_out[m,n] = C + bias[n]                                  (x@W_ih^T + b_ih)
// mode 1: a = tanh(C + aux[m*stride+n] + bias[n]); h_out=a; u_out=a+hnext   (layer 0)
// mode 2: a = tanh(C + 2*bias[n]);                 h_out=a; u_out=a+hnext   (layers 1..4)
// mode 3: a = tanh(C + 2*bias[n]);                 h_out=a; seq_t[m*stride+n]=a (layer 5)
//
// 128 threads = 4 waves arranged 2(M)x2(N); each wave owns a 32x32 tile =
// 2x2 WMMA accumulators -> 8 b128 loads per 4 v_wmma (2 loads/WMMA) with
// register-level A/B fragment reuse. Grid: (M/64, N/64).
__global__ void __launch_bounds__(128) gemm_rnn(
    const __bf16* __restrict__ A, const __bf16* __restrict__ W,
    int N, int K, int mode,
    const float* __restrict__ bias,
    const float* __restrict__ aux_f32, long aux_mstride,
    __bf16* __restrict__ h_out,
    __bf16* __restrict__ u_out,
    const __bf16* __restrict__ hnext,
    __bf16* __restrict__ seq_t,
    float* __restrict__ f32_out)
{
    const int lane = threadIdx.x & 31;
    const int wave = threadIdx.x >> 5;                 // 0..3
    const int m0 = blockIdx.x * 64 + (wave & 1) * 32;
    const int n0 = blockIdx.y * 64 + (wave >> 1) * 32;

    // A operand (16-bit 16x32): lane%16 = M row; lanes<16 hold K {0..7,16..23},
    // lanes>=16 hold K {8..15,24..31}.
    const int am    = m0 + (lane & 15);
    const int kaoff = (lane >> 4) * 8;
    // B operand (32x16): lane%16 = N col; lanes<16 hold K {0..7,16..23},
    // lanes>=16 hold K {8..15,24..31}.
    const int bn    = n0 + (lane & 15);
    const int kboff = (lane >> 4) * 8;

    const __bf16* aptr0 = A + (size_t)am * K + kaoff;
    const __bf16* aptr1 = aptr0 + (size_t)16 * K;      // rows m0+16..m0+31
    const __bf16* bptr0 = W + (size_t)bn * K + kboff;
    const __bf16* bptr1 = bptr0 + (size_t)16 * K;      // cols n0+16..n0+31

    v8f c00 = {}, c01 = {}, c10 = {}, c11 = {};
    for (int kk = 0; kk < K; kk += 32) {
        v16bf a0 = cat16(*(const bf16x8*)(aptr0 + kk), *(const bf16x8*)(aptr0 + kk + 16));
        v16bf a1 = cat16(*(const bf16x8*)(aptr1 + kk), *(const bf16x8*)(aptr1 + kk + 16));
        v16bf b0 = cat16(*(const bf16x8*)(bptr0 + kk), *(const bf16x8*)(bptr0 + kk + 16));
        v16bf b1 = cat16(*(const bf16x8*)(bptr1 + kk), *(const bf16x8*)(bptr1 + kk + 16));
        // (neg_a, A, neg_b, B, c_mod, C, reuse_a, reuse_b)
        c00 = __builtin_amdgcn_wmma_f32_16x16x32_bf16(false, a0, false, b0, (short)0, c00, false, false);
        c01 = __builtin_amdgcn_wmma_f32_16x16x32_bf16(false, a0, false, b1, (short)0, c01, false, false);
        c10 = __builtin_amdgcn_wmma_f32_16x16x32_bf16(false, a1, false, b0, (short)0, c10, false, false);
        c11 = __builtin_amdgcn_wmma_f32_16x16x32_bf16(false, a1, false, b1, (short)0, c11, false, false);
    }

    // C/D layout: VGPR v, lane L -> M = v + 8*(L>=16), N = L%16 (per 16x16 tile)
    const int nl = lane & 15;
    const int mh = (lane >> 4) ? 8 : 0;

    auto epi = [&](const v8f& cc, int dm, int dn) {
        const int n     = n0 + dn + nl;
        const int mbase = m0 + dm + mh;
        const float bnf = bias[n];
        if (mode == 0) {
#pragma unroll
            for (int v = 0; v < 8; ++v)
                f32_out[(size_t)(mbase + v) * N + n] = cc[v] + bnf;
            return;
        }
#pragma unroll
        for (int v = 0; v < 8; ++v) {
            const int m = mbase + v;
            float pre = cc[v];
            if (mode == 1) pre += aux_f32[(size_t)m * aux_mstride + n] + bnf;
            else           pre += 2.0f * bnf;
            const float a = tanhf(pre);
            h_out[(size_t)m * N + n] = f2bf(a);
            if (mode != 3)
                u_out[(size_t)m * N + n] = f2bf(a + bf2f(hnext[(size_t)m * N + n]));
            else
                seq_t[(size_t)m * aux_mstride + n] = f2bf(a);
        }
    };
    epi(c00, 0, 0); epi(c01, 0, 16); epi(c10, 16, 0); epi(c11, 16, 16);
}

// ---- output projection: out[b*S+t] = seq[b,t,:] . w_out + b_out ---------
__global__ void __launch_bounds__(256) proj_out(
    const __bf16* __restrict__ seq, const float* __restrict__ wout,
    const float* __restrict__ bout, float* __restrict__ out, int rows, int Hn)
{
    const int gw   = (blockIdx.x * blockDim.x + threadIdx.x) >> 5;
    const int lane = threadIdx.x & 31;
    if (gw >= rows) return;
    const __bf16* row = seq + (size_t)gw * Hn;
    float s = 0.f;
    for (int i = lane; i < Hn; i += 32) s += bf2f(row[i]) * wout[i];
#pragma unroll
    for (int off = 16; off; off >>= 1) s += __shfl_xor(s, off, 32);
    if (lane == 0) out[gw] = s + bout[0];
}

// ---- h_final bf16 -> f32 -------------------------------------------------
__global__ void hfinal_f32(const __bf16* __restrict__ h, float* __restrict__ out, int n) {
    for (int i = blockIdx.x * blockDim.x + threadIdx.x; i < n; i += gridDim.x * blockDim.x)
        out[i] = bf2f(h[i]);
}

extern "C" void kernel_launch(void* const* d_in, const int* in_sizes, int n_in,
                              void* d_out, int out_size, void* d_ws, size_t ws_size,
                              hipStream_t stream)
{
    const float* x    = (const float*)d_in[0];
    const float* Wih  = (const float*)d_in[1];
    const float* bih  = (const float*)d_in[2];
    const float* Whh  = (const float*)d_in[3];
    const float* bhh  = (const float*)d_in[4];
    const float* wout = (const float*)d_in[5];
    const float* bout = (const float*)d_in[6];
    float* out = (float*)d_out;
    (void)in_sizes; (void)n_in; (void)out_size; (void)ws_size;

    // workspace carve-out (~240 MB)
    char* ws = (char*)d_ws;
    size_t off = 0;
    auto alloc = [&](size_t bytes) -> char* {
        char* p = ws + off;
        off = (off + bytes + 255) & ~(size_t)255;
        return p;
    };
    __bf16* xbf  = (__bf16*)alloc((size_t)B_ * S_ * I_ * 2);   // x in bf16
    __bf16* wihb = (__bf16*)alloc((size_t)H_ * I_ * 2);        // W_ih bf16
    __bf16* whhb = (__bf16*)alloc((size_t)H_ * H_ * 2);        // W_hh bf16
    float*  XW   = (float*) alloc((size_t)B_ * S_ * H_ * 4);   // x@W_ih^T + b_ih (f32)
    __bf16* h    = (__bf16*)alloc((size_t)L_ * B_ * H_ * 2);   // hidden state [L][B][H]
    __bf16* u0   = (__bf16*)alloc((size_t)B_ * H_ * 2);        // fused (a+h_next) ping
    __bf16* u1   = (__bf16*)alloc((size_t)B_ * H_ * 2);        // fused (a+h_next) pong
    __bf16* seq  = (__bf16*)alloc((size_t)B_ * S_ * H_ * 2);   // top-layer outputs

    // 1) precision conversion + state init
    cvt_f32_bf16<<<4096, 256, 0, stream>>>(x,   xbf,  B_ * S_ * I_);
    cvt_f32_bf16<<<1024, 256, 0, stream>>>(Wih, wihb, H_ * I_);
    cvt_f32_bf16<<<2048, 256, 0, stream>>>(Whh, whhb, H_ * H_);
    zero_u32<<<512, 256, 0, stream>>>((unsigned*)h, L_ * B_ * H_ / 2);

    // 2) hoisted input projection: XW[b*S+t][n] = x . W_ih^T + b_ih (one big GEMM)
    {
        dim3 g(B_ * S_ / 64, H_ / 64);
        gemm_rnn<<<g, 128, 0, stream>>>(xbf, wihb, H_, I_, 0, bih,
                                        nullptr, 0, nullptr, nullptr, nullptr,
                                        nullptr, XW);
    }

    // 3) sequential recurrence: 6 fused GEMMs per timestep (L2-resident)
    const size_t BH = (size_t)B_ * H_;
    const dim3 gl(1, H_ / 64);
    for (int t = 0; t < S_; ++t) {
        // layer 0: a0 = tanh(h0@W^T + XW[:,t,:] + b); u0 = a0 + h1_old
        gemm_rnn<<<gl, 128, 0, stream>>>(h, whhb, H_, H_, 1, bhh,
                                         XW + (size_t)t * H_, (long)S_ * H_,
                                         h, u0, h + BH, nullptr, nullptr);
        // layers 1..4: a = tanh(u@W^T + 2b); u_next = a + h_{l+1}_old
        for (int l = 1; l <= 4; ++l) {
            __bf16* uin  = (l & 1) ? u0 : u1;
            __bf16* uout = (l & 1) ? u1 : u0;
            gemm_rnn<<<gl, 128, 0, stream>>>(uin, whhb, H_, H_, 2, bhh,
                                             nullptr, 0,
                                             h + (size_t)l * BH, uout,
                                             h + (size_t)(l + 1) * BH,
                                             nullptr, nullptr);
        }
        // layer 5: a5 = tanh(u@W^T + 2b); emit to seq[:,t,:]
        gemm_rnn<<<gl, 128, 0, stream>>>(u0, whhb, H_, H_, 3, bhh,
                                         nullptr, (long)S_ * H_,
                                         h + 5 * BH, nullptr, nullptr,
                                         seq + (size_t)t * H_, nullptr);
    }

    // 4) outputs: projection [B,S,1] then h_final [L,B,H] (concatenated in d_out)
    proj_out<<<(B_ * S_ * 32 + 255) / 256, 256, 0, stream>>>(seq, wout, bout,
                                                             out, B_ * S_, H_);
    hfinal_f32<<<1536, 256, 0, stream>>>(h, out + B_ * S_, L_ * B_ * H_);
}
